// Varkeys_52888227283656
// MI455X (gfx1250) — matrix operand order repaired and checked by reading, not compiled, not run on hardware
//
#include <hip/hip_runtime.h>
#include <hip/hip_bf16.h>

// ---------------------------------------------------------------------------
// Fused reciprocal-distance kernel attention for MI455X (gfx1250, wave32).
//   d = (|k|^2 - 2 k.x + |x|^2)/SCALE ; ker = 1/(d+EPS) ; out = rownorm(ker^T V)
// Chained-GEMM fusion (never materialize ker), bf16 WMMA w/ f32 accumulate.
// Global loads for chunk i+1 issued before compute of chunk i (reg pipeline);
// fragment loads batched so WMMAs issue back-to-back; hardware v_rcp_f32 for
// the kernel reciprocal (result is truncated to bf16 anyway).
// ---------------------------------------------------------------------------

typedef __attribute__((ext_vector_type(16))) __bf16    v16bf;
typedef __attribute__((ext_vector_type(8)))  float     v8f;
typedef __attribute__((ext_vector_type(4)))  unsigned  uv4;
typedef __attribute__((ext_vector_type(4)))  float     fv4;

#define D_KEYS   50000
#define B_ROWS   4096
#define KDIM     64
#define NCAT     1000
#define SCALE_INV (1.0f / 300000.0f)
#define EPS_K    1e-4f

#define BM 256          // x-rows per block (16 waves x 16 rows)
#define BN 128          // categories per block
#define BD 64           // d-chunk depth
#define NWAVES 16
#define NTHREADS 512
#define LDS_STRIDE 144  // 64 bf16 (128B) + 16B pad -> conflict-free b128 reads

__device__ __forceinline__ unsigned pack2bf(float a, float b) {
  unsigned ua = __float_as_uint(a), ub = __float_as_uint(b);
  ua = (ua + 0x7FFFu + ((ua >> 16) & 1u)) >> 16;
  ub = (ub + 0x7FFFu + ((ub >> 16) & 1u)) >> 16;
  return (ub << 16) | (ua & 0xFFFFu);
}
__device__ __forceinline__ unsigned short bf16of(float a) {
  unsigned ua = __float_as_uint(a);
  return (unsigned short)((ua + 0x7FFFu + ((ua >> 16) & 1u)) >> 16);
}

union Frag {            // 8 VGPRs: one 16x32 bf16 A-frag or 32x16 bf16 B-frag
  v16bf v;
  uv4   q[2];
  unsigned u[8];
};

#define WMMA_BF16(A, B, C) \
  __builtin_amdgcn_wmma_f32_16x16x32_bf16(false, (A), false, (B), (short)0, (C), false, false)

// --------------------------- prepass: row norms -----------------------------
__global__ __launch_bounds__(256) void row_norms_kernel(
    const float* __restrict__ keys, const float* __restrict__ x,
    float* __restrict__ knorm, float* __restrict__ xnorm) {
  int gw   = (blockIdx.x * 256 + threadIdx.x) >> 5;   // one wave per row
  int lane = threadIdx.x & 31;
  float s;
  if (gw < D_KEYS) {
    const float* r = keys + (size_t)gw * KDIM;
    float a = r[lane], b = r[lane + 32];
    s = fmaf(a, a, b * b);
  } else if (gw < D_KEYS + B_ROWS) {
    const float* r = x + (size_t)(gw - D_KEYS) * KDIM;
    float a = r[lane], b = r[lane + 32];
    s = fmaf(a, a, b * b);
  } else {
    return;
  }
#pragma unroll
  for (int o = 16; o > 0; o >>= 1) s += __shfl_xor(s, o, 32);
  if (lane == 0) {
    if (gw < D_KEYS) knorm[gw] = s;
    else             xnorm[gw - D_KEYS] = s;
  }
}

// --------------------------- fused main kernel ------------------------------
__global__ __launch_bounds__(NTHREADS) void varkeys_fused_kernel(
    const float* __restrict__ x, const float* __restrict__ keys,
    const float* __restrict__ V, const float* __restrict__ knorm,
    const float* __restrict__ xnorm, float* __restrict__ out) {
  // Separate LDS arrays (helps alias analysis keep ker stores independent of
  // V-fragment loads).  keys chunk [64 d][64 k] bf16 ; V chunk transposed
  // [128 n][64 d] bf16 ; per-wave ker tile [16 m][64 d] bf16. Rows strided 144B.
  __shared__ unsigned char keys_sm[64 * LDS_STRIDE];
  __shared__ unsigned char v_sm[128 * LDS_STRIDE];
  __shared__ unsigned char ker_sm[NWAVES * 16 * LDS_STRIDE];

  const int tid  = threadIdx.x;
  const int lane = tid & 31;
  const int wave = tid >> 5;
  const int nl   = lane & 15;      // N/M index within half-wave
  const int hl   = lane >> 4;      // half-wave select
  const int mblk = blockIdx.x * BM;
  const int nblk = blockIdx.y * BN;
  const int mwave = mblk + wave * 16;   // this wave's 16 x-rows

  // staging geometry (fixed per thread)
  const int sk_r  = tid >> 3;           // keys: d row 0..63
  const int sk_kc = (tid & 7) * 8;      // keys: k base 0..56
  int sv_p[2], sv_q[2];
#pragma unroll
  for (int u = 0; u < 2; ++u) {
    int unit = u * NTHREADS + tid;      // 1024 units: 32 d-pairs x 32 n-quads
    sv_p[u] = unit >> 5;
    sv_q[u] = unit & 31;
  }

  // ---- preload x rows as bf16 A-fragments (K=0..31, K=32..63), once ----
  Frag ax0, ax1;
  {
    const float* xrow = x + (size_t)(mwave + nl) * KDIM;
#pragma unroll
    for (int f = 0; f < 2; ++f) {
      int kb = f * 32 + hl * 16;
      const fv4* p0 = (const fv4*)(xrow + kb);
      const fv4* p1 = (const fv4*)(xrow + kb + 8);
      fv4 lo0 = p0[0], lo1 = p0[1], hi0 = p1[0], hi1 = p1[1];
      Frag& A = f ? ax1 : ax0;
      A.u[0] = pack2bf(lo0.x, lo0.y); A.u[1] = pack2bf(lo0.z, lo0.w);
      A.u[2] = pack2bf(lo1.x, lo1.y); A.u[3] = pack2bf(lo1.z, lo1.w);
      A.u[4] = pack2bf(hi0.x, hi0.y); A.u[5] = pack2bf(hi0.z, hi0.w);
      A.u[6] = pack2bf(hi1.x, hi1.y); A.u[7] = pack2bf(hi1.z, hi1.w);
    }
  }
  // xnorm for the 8 M values this lane owns in a C tile (m = 8*hl + i)
  float xn[8];
#pragma unroll
  for (int i = 0; i < 8; ++i) xn[i] = xnorm[mwave + 8 * hl + i];

  const fv4 fz = {0.f, 0.f, 0.f, 0.f};
  const v8f vzero = {0.f, 0.f, 0.f, 0.f, 0.f, 0.f, 0.f, 0.f};
  v8f acc[8];
#pragma unroll
  for (int j = 0; j < 8; ++j) acc[j] = vzero;

  unsigned char* kerw = ker_sm + wave * (16 * LDS_STRIDE);

  // pipeline registers for next chunk's global data
  fv4 gk0, gk1;         // keys: 8 floats
  fv4 gva[2], gvb[2];   // V: 2 units x (row d0, row d0+1) float4

  auto load_stage = [&](int dStart) {
    int dg = dStart + sk_r;
    gk0 = fz; gk1 = fz;
    if (dg < D_KEYS) {
      const fv4* kp = (const fv4*)(keys + (size_t)dg * KDIM + sk_kc);
      gk0 = kp[0]; gk1 = kp[1];
      if (dg + BD < D_KEYS)
        __builtin_prefetch(keys + (size_t)(dg + BD) * KDIM + sk_kc, 0, 0);
    }
#pragma unroll
    for (int u = 0; u < 2; ++u) {
      int d0 = dStart + 2 * sv_p[u];
      int n0 = nblk + 4 * sv_q[u];
      gva[u] = fz; gvb[u] = fz;
      if (n0 < NCAT) {
        if (d0 < D_KEYS)     gva[u] = *(const fv4*)(V + (size_t)d0 * NCAT + n0);
        if (d0 + 1 < D_KEYS) gvb[u] = *(const fv4*)(V + (size_t)(d0 + 1) * NCAT + n0);
        if (d0 + BD < D_KEYS)
          __builtin_prefetch(V + (size_t)(d0 + BD) * NCAT + n0, 0, 0);
      }
    }
  };

  auto store_stage = [&]() {
    uv4 pk;
    pk.x = pack2bf(gk0.x, gk0.y); pk.y = pack2bf(gk0.z, gk0.w);
    pk.z = pack2bf(gk1.x, gk1.y); pk.w = pack2bf(gk1.z, gk1.w);
    *(uv4*)(keys_sm + sk_r * LDS_STRIDE + sk_kc * 2) = pk;
#pragma unroll
    for (int u = 0; u < 2; ++u) {
      unsigned char* dst = v_sm + (4 * sv_q[u]) * LDS_STRIDE + sv_p[u] * 4;
      *(unsigned*)(dst + 0 * LDS_STRIDE) = pack2bf(gva[u].x, gvb[u].x);
      *(unsigned*)(dst + 1 * LDS_STRIDE) = pack2bf(gva[u].y, gvb[u].y);
      *(unsigned*)(dst + 2 * LDS_STRIDE) = pack2bf(gva[u].z, gvb[u].z);
      *(unsigned*)(dst + 3 * LDS_STRIDE) = pack2bf(gva[u].w, gvb[u].w);
    }
  };

  // prologue: stage chunk 0
  load_stage(0);
  store_stage();
  __syncthreads();

  for (int dc = 0; dc < D_KEYS; dc += BD) {
    // issue next chunk's global loads now; they complete during compute
    load_stage(dc + BD);

    // ---- GEMM1: batched B-frag loads, then back-to-back WMMAs ----
    Frag kb[8];
#pragma unroll
    for (int t = 0; t < 4; ++t) {
      const unsigned char* kp = keys_sm + (16 * t + nl) * LDS_STRIDE + hl * 32;
      kb[2 * t + 0].q[0] = *(const uv4*)(kp);
      kb[2 * t + 0].q[1] = *(const uv4*)(kp + 16);
      kb[2 * t + 1].q[0] = *(const uv4*)(kp + 64);
      kb[2 * t + 1].q[1] = *(const uv4*)(kp + 80);
    }
    // hoist knorm loads (VMEM) ahead of the math that consumes them
    float kn[4];
#pragma unroll
    for (int t = 0; t < 4; ++t) {
      int dg = dc + 16 * t + nl;
      kn[t] = (dg < D_KEYS) ? knorm[dg] : 0.0f;
    }
    v8f c1[4];
#pragma unroll
    for (int t = 0; t < 4; ++t) c1[t] = vzero;
#pragma unroll
    for (int t = 0; t < 4; ++t) {
      c1[t] = WMMA_BF16(ax0.v, kb[2 * t + 0].v, c1[t]);
      c1[t] = WMMA_BF16(ax1.v, kb[2 * t + 1].v, c1[t]);
    }

    // ---- epilogue: ker = rcp((xn + kn - 2S)/SCALE + EPS) -> ker_sm [m][d] ----
    // v_rcp_f32 (~1 ulp) is plenty: result is truncated to bf16 (8 mantissa
    // bits) before GEMM2.  Avoids the IEEE div_scale/Newton/div_fmas chain.
#pragma unroll
    for (int t = 0; t < 4; ++t) {
      int dloc = 16 * t + nl;
      int dg   = dc + dloc;
#pragma unroll
      for (int i = 0; i < 8; ++i) {
        float S  = c1[t][i];
        float dv = fmaf(S, -2.0f * SCALE_INV, fmaf(xn[i] + kn[t], SCALE_INV, EPS_K));
        float kv = (dg < D_KEYS) ? __builtin_amdgcn_rcpf(dv) : 0.0f;
        *(unsigned short*)(kerw + (8 * hl + i) * LDS_STRIDE + dloc * 2) = bf16of(kv);
      }
    }

    // ---- reload ker as A-frags (wave-private LDS; DS in-order) ----
    Frag ka0, ka1;
    {
      const unsigned char* ka = kerw + nl * LDS_STRIDE + hl * 32;
      ka0.q[0] = *(const uv4*)(ka);       ka0.q[1] = *(const uv4*)(ka + 16);
      ka1.q[0] = *(const uv4*)(ka + 64);  ka1.q[1] = *(const uv4*)(ka + 80);
    }

    // ---- GEMM2: two halves of (8 batched frag loads -> 8 WMMAs) ----
#pragma unroll
    for (int half = 0; half < 2; ++half) {
      Frag vb[8];
#pragma unroll
      for (int jj = 0; jj < 4; ++jj) {
        int j = half * 4 + jj;
        const unsigned char* vp = v_sm + (16 * j + nl) * LDS_STRIDE + hl * 32;
        vb[2 * jj + 0].q[0] = *(const uv4*)(vp);
        vb[2 * jj + 0].q[1] = *(const uv4*)(vp + 16);
        vb[2 * jj + 1].q[0] = *(const uv4*)(vp + 64);
        vb[2 * jj + 1].q[1] = *(const uv4*)(vp + 80);
      }
#pragma unroll
      for (int jj = 0; jj < 4; ++jj) {
        int j = half * 4 + jj;
        acc[j] = WMMA_BF16(ka0.v, vb[2 * jj + 0].v, acc[j]);
        acc[j] = WMMA_BF16(ka1.v, vb[2 * jj + 1].v, acc[j]);
      }
    }

    __syncthreads();   // all waves done reading LDS for chunk dc
    store_stage();     // pack + ds_store next chunk (registers -> LDS)
    __syncthreads();   // stores visible to all waves
  }

  // ---- write KV tile (normalized by a follow-up pass) ----
#pragma unroll
  for (int j = 0; j < 8; ++j) {
    int n = nblk + 16 * j + nl;
    if (n < NCAT) {
#pragma unroll
      for (int i = 0; i < 8; ++i)
        out[(size_t)(mwave + 8 * hl + i) * NCAT + n] = acc[j][i];
    }
  }
}

// --------------------------- row normalization ------------------------------
__global__ __launch_bounds__(256) void row_normalize_kernel(float* __restrict__ out) {
  int row = blockIdx.x;
  float s = 0.f;
  for (int c = threadIdx.x; c < NCAT; c += 256) s += out[(size_t)row * NCAT + c];
#pragma unroll
  for (int o = 16; o > 0; o >>= 1) s += __shfl_xor(s, o, 32);
  __shared__ float red[8];
  int lane = threadIdx.x & 31, wv = threadIdx.x >> 5;
  if (lane == 0) red[wv] = s;
  __syncthreads();
  float tot = 0.f;
#pragma unroll
  for (int w = 0; w < 8; ++w) tot += red[w];
  float inv = 1.0f / tot;
  for (int c = threadIdx.x; c < NCAT; c += 256) out[(size_t)row * NCAT + c] *= inv;
}

// ---------------------------------------------------------------------------
extern "C" void kernel_launch(void* const* d_in, const int* in_sizes, int n_in,
                              void* d_out, int out_size, void* d_ws, size_t ws_size,
                              hipStream_t stream) {
  const float* x    = (const float*)d_in[0];   // (4096, 64)
  const float* keys = (const float*)d_in[1];   // (50000, 64)
  const float* V    = (const float*)d_in[2];   // (50000, 1000)
  float* out = (float*)d_out;                  // (4096, 1000)

  float* knorm = (float*)d_ws;                 // 50000 (padded to 50048)
  float* xnorm = knorm + 50048;                // 4096

  // 54096 rows, one wave each, 8 waves per 256-thread block
  row_norms_kernel<<<(D_KEYS + B_ROWS) / 8, 256, 0, stream>>>(keys, x, knorm, xnorm);

  dim3 grid(B_ROWS / BM, (NCAT + BN - 1) / BN);   // (16, 8)
  varkeys_fused_kernel<<<grid, NTHREADS, 0, stream>>>(x, keys, V, knorm, xnorm, out);

  row_normalize_kernel<<<B_ROWS, 256, 0, stream>>>(out);
}
